// ResidualVectorQuantizer_57681410785797
// MI455X (gfx1250) — compile-verified, matrix operand order checked by reading, not run on hardware
//
#include <hip/hip_runtime.h>
#include <stdint.h>

typedef _Float16 v16h __attribute__((ext_vector_type(16)));
typedef _Float16 v8h  __attribute__((ext_vector_type(8)));
typedef float    v8f  __attribute__((ext_vector_type(8)));

#define NQ      4
#define KCB     1024
#define DIM     256
#define NROW    (16*8192)          // 131072 rows
#define QELEMS  (NROW*DIM)         // 33554432 floats (quantized_out)
#define IELEMS  (NQ*NROW)          // 524288 floats (indices, stored as float)
#define ROWS    128                // rows per workgroup (8 waves x 16)
#define BLOCK   256
#define CTILES  (KCB/16)           // 64 column tiles of 16 codewords
#define TILE_BYTES (16*DIM*2)      // 8192 bytes per f16 B tile

// ws layout (floats): [0] loss accum | [64..4160) codebook norms | byte 32768: f16 codebook (2MB)
#define WS_EN_OFF   64
#define WS_EMBH_OFF 32768          // byte offset

// ---------------------------------------------------------------- init: zero qout + loss accumulator
__global__ __launch_bounds__(256) void rvq_init(float* __restrict__ qout, float* __restrict__ loss) {
  size_t i = ((size_t)blockIdx.x * 256 + threadIdx.x) * 4;
  float4 z = {0.f, 0.f, 0.f, 0.f};
  *(float4*)(qout + i) = z;
  if (blockIdx.x == 0 && threadIdx.x == 0) *loss = 0.f;
}

// ---------------------------------------------------------------- prep: codebook norms + f16 codebook
__global__ __launch_bounds__(256) void rvq_prep(const float* __restrict__ emb,
                                                float* __restrict__ en,
                                                _Float16* __restrict__ embH) {
  int c = blockIdx.x * 256 + threadIdx.x;       // 0..4095 (layer*1024 + codeword)
  const float* e = emb + (size_t)c * DIM;
  _Float16*    d = embH + (size_t)c * DIM;
  float s = 0.f;
  for (int k = 0; k < DIM; k += 8) {
    float4 va = *(const float4*)(e + k);
    float4 vb = *(const float4*)(e + k + 4);
    s = fmaf(va.x, va.x, s); s = fmaf(va.y, va.y, s);
    s = fmaf(va.z, va.z, s); s = fmaf(va.w, va.w, s);
    s = fmaf(vb.x, vb.x, s); s = fmaf(vb.y, vb.y, s);
    s = fmaf(vb.z, vb.z, s); s = fmaf(vb.w, vb.w, s);
    v8h h;
    h[0] = (_Float16)va.x; h[1] = (_Float16)va.y; h[2] = (_Float16)va.z; h[3] = (_Float16)va.w;
    h[4] = (_Float16)vb.x; h[5] = (_Float16)vb.y; h[6] = (_Float16)vb.z; h[7] = (_Float16)vb.w;
    *(v8h*)(d + k) = h;
  }
  en[c] = s;
}

// issue one 8KB B-tile copy (global f16 -> LDS) via async-to-LDS, 2 x b128 per thread
__device__ __forceinline__ void issue_btile(uint32_t lds_base, uint64_t gbase, int tid) {
  uint32_t loff = lds_base + (uint32_t)tid * 16u;
  uint64_t g    = gbase + (uint64_t)tid * 16u;
  asm volatile("global_load_async_to_lds_b128 %0, %1, off"
               :: "v"(loff), "v"(g) : "memory");
  asm volatile("global_load_async_to_lds_b128 %0, %1, off offset:4096"
               :: "v"(loff), "v"(g) : "memory");
}

// ---------------------------------------------------------------- one RVQ layer (GEMM + argmin + update)
__global__ __launch_bounds__(BLOCK) void rvq_layer(
    const float* __restrict__ x, const float* __restrict__ emb,
    const _Float16* __restrict__ embH,
    float* __restrict__ qout, float* __restrict__ idx_out,
    const float* __restrict__ en, float* __restrict__ loss, int layer) {

  __shared__ __align__(16) _Float16 Bh[2 * 16 * DIM];   // double-buffered f16 B tile (2 x 8KB)
  __shared__ int   idxT[ROWS];
  __shared__ float sred[BLOCK];

  const int  tid  = threadIdx.x;
  const int  wave = tid >> 5;
  const int  lane = tid & 31;
  const int  mrow = lane & 15;
  const int  hi   = lane >> 4;
  const long base = (long)blockIdx.x * ROWS;
  const float*    embL  = emb  + (size_t)layer * KCB * DIM;
  const _Float16* embHL = embH + (size_t)layer * KCB * DIM;
  const float*    enL   = en   + layer * KCB;

  const uint32_t ldsB  = (uint32_t)(size_t)(&Bh[0]);
  const uint64_t gB    = (uint64_t)(size_t)(const void*)embHL;

  // ---- Build A fragments once: residual = x - qout, f32 -> f16, WMMA 16x32 A layout.
  v16h afrag[8];
  {
    const long   arow = base + wave * 16 + mrow;
    const float* xr = x    + arow * DIM;
    const float* qr = qout + arow * DIM;
#pragma unroll
    for (int kk = 0; kk < 8; ++kk) {
      const int kb = kk * 32 + hi * 8;
      float4 x0 = *(const float4*)(xr + kb);
      float4 x1 = *(const float4*)(xr + kb + 4);
      float4 x2 = *(const float4*)(xr + kb + 16);
      float4 x3 = *(const float4*)(xr + kb + 20);
      float4 q0 = *(const float4*)(qr + kb);
      float4 q1 = *(const float4*)(qr + kb + 4);
      float4 q2 = *(const float4*)(qr + kb + 16);
      float4 q3 = *(const float4*)(qr + kb + 20);
      v16h a;
      a[0]  = (_Float16)(x0.x - q0.x); a[1]  = (_Float16)(x0.y - q0.y);
      a[2]  = (_Float16)(x0.z - q0.z); a[3]  = (_Float16)(x0.w - q0.w);
      a[4]  = (_Float16)(x1.x - q1.x); a[5]  = (_Float16)(x1.y - q1.y);
      a[6]  = (_Float16)(x1.z - q1.z); a[7]  = (_Float16)(x1.w - q1.w);
      a[8]  = (_Float16)(x2.x - q2.x); a[9]  = (_Float16)(x2.y - q2.y);
      a[10] = (_Float16)(x2.z - q2.z); a[11] = (_Float16)(x2.w - q2.w);
      a[12] = (_Float16)(x3.x - q3.x); a[13] = (_Float16)(x3.y - q3.y);
      a[14] = (_Float16)(x3.z - q3.z); a[15] = (_Float16)(x3.w - q3.w);
      afrag[kk] = a;
    }
  }

  float best_s[8];
  int   best_i[8];
#pragma unroll
  for (int j = 0; j < 8; ++j) { best_s[j] = 3.4e38f; best_i[j] = 0; }

  // prologue: async-stage tile 0 into buffer 0
  issue_btile(ldsB, gB, tid);

  for (int ct = 0; ct < CTILES; ++ct) {
    // my wave's async loads for tile ct have landed in LDS
    asm volatile("s_wait_asynccnt 0x0" ::: "memory");
    __syncthreads();   // all waves' portions visible; previous buffer fully consumed

    if (ct + 1 < CTILES) {
      // overlap: stream tile ct+1 into the other buffer while we compute tile ct
      issue_btile(ldsB + (uint32_t)((ct + 1) & 1) * TILE_BYTES,
                  gB + (uint64_t)(ct + 1) * TILE_BYTES, tid);
      __builtin_prefetch(embHL + (size_t)((ct + 2 < CTILES ? ct + 2 : ct + 1) * 16 + mrow) * DIM, 0, 1);
    }

    const _Float16* Bt = Bh + (ct & 1) * 16 * DIM;

    // ---- 16x16 score tile: S = R * E^T, 8 chained K-steps of 32
    v8f c = {};
#pragma unroll
    for (int kk = 0; kk < 8; ++kk) {
      const int kb = kk * 32 + hi * 8;
      v8h b0 = *(const v8h*)(Bt + mrow * DIM + kb);
      v8h b1 = *(const v8h*)(Bt + mrow * DIM + kb + 16);
      v16h b;
#pragma unroll
      for (int i = 0; i < 8; ++i) { b[i] = b0[i]; b[i + 8] = b1[i]; }
      c = __builtin_amdgcn_wmma_f32_16x16x32_f16(false, afrag[kk], false, b,
                                                 (short)0, c, false, false);
    }

    // ---- running argmin: score = ||e||^2 - 2*dot  (lane = column n, c[j] = row)
    const int   n   = ct * 16 + mrow;
    const float enk = enL[n];
#pragma unroll
    for (int j = 0; j < 8; ++j) {
      float s = fmaf(-2.f, c[j], enk);
      if (s < best_s[j]) { best_s[j] = s; best_i[j] = n; }
    }
  }

  // ---- reduce argmin over the 16 columns held by each 16-lane group (lower-index wins ties)
#pragma unroll
  for (int off = 1; off < 16; off <<= 1) {
#pragma unroll
    for (int j = 0; j < 8; ++j) {
      float os = __shfl_xor(best_s[j], off, 16);
      int   oi = __shfl_xor(best_i[j], off, 16);
      if (os < best_s[j] || (os == best_s[j] && oi < best_i[j])) {
        best_s[j] = os; best_i[j] = oi;
      }
    }
  }
  if (mrow == 0) {   // lane 0 -> rows 0..7 of this wave's tile, lane 16 -> rows 8..15
#pragma unroll
    for (int j = 0; j < 8; ++j) idxT[wave * 16 + hi * 8 + j] = best_i[j];
  }
  __syncthreads();

  if (tid < ROWS) idx_out[base + tid] = (float)idxT[tid];

  // ---- gather codewords (f32), update qout, accumulate ||new residual||^2
  float lsq = 0.f;
  for (int e = tid; e < ROWS * DIM; e += BLOCK) {
    const int   row = e >> 8;
    const int   k   = e & 255;
    const long  g   = base * DIM + e;
    const float q   = embL[(size_t)idxT[row] * DIM + k];
    const float qo  = qout[g];
    const float rn  = x[g] - qo - q;   // residual AFTER this layer; (q - r_before)^2 == rn^2
    qout[g] = qo + q;
    lsq += rn * rn;
  }
  sred[tid] = lsq;
  __syncthreads();
  for (int s = BLOCK / 2; s > 0; s >>= 1) {
    if (tid < s) sred[tid] += sred[tid + s];
    __syncthreads();
  }
  if (tid == 0) atomicAdd(loss, sred[0]);
}

// ---------------------------------------------------------------- finalize scalar losses
__global__ void rvq_final(const float* __restrict__ loss, float* __restrict__ out) {
  if (threadIdx.x == 0 && blockIdx.x == 0) {
    float m = *loss * (1.0f / (float)QELEMS);  // sum over layers of per-layer MSE
    out[0] = 1.25f * m;   // vq_loss
    out[1] = 0.25f * m;   // commit_loss
    out[2] = m;           // codebook_loss
  }
}

extern "C" void kernel_launch(void* const* d_in, const int* in_sizes, int n_in,
                              void* d_out, int out_size, void* d_ws, size_t ws_size,
                              hipStream_t stream) {
  const float* x   = (const float*)d_in[0];   // (16, 8192, 256) f32
  const float* emb = (const float*)d_in[1];   // (4, 1024, 256) f32
  float* out     = (float*)d_out;
  float* qout    = out;                       // [0, QELEMS)
  float* idx0    = out + QELEMS;              // [QELEMS, QELEMS+IELEMS) indices as float
  float* lossout = out + QELEMS + IELEMS;     // 3 scalars
  float* wsf     = (float*)d_ws;
  float* loss    = wsf;                       // ws[0]: squared-error accumulator
  float* en      = wsf + WS_EN_OFF;           // 4096 floats: codebook norms
  _Float16* embH = (_Float16*)((char*)d_ws + WS_EMBH_OFF);  // 4*1024*256 f16 codebook

  rvq_init<<<QELEMS / (256 * 4), 256, 0, stream>>>(qout, loss);
  rvq_prep<<<(NQ * KCB) / 256, 256, 0, stream>>>(emb, en, embH);
  for (int l = 0; l < NQ; ++l)
    rvq_layer<<<NROW / ROWS, BLOCK, 0, stream>>>(x, emb, embH, qout,
                                                 idx0 + (size_t)l * NROW, en, loss, l);
  rvq_final<<<1, 1, 0, stream>>>(loss, lossout);
}
